// Transformer_2388001817316
// MI455X (gfx1250) — matrix operand order repaired and checked by reading, not compiled, not run on hardware
//
#include <hip/hip_runtime.h>
#include <hip/hip_bf16.h>
#include <math.h>

#define NTOK   2304   // SEQ * (r+1) = 768 * 3
#define DMODEL 512
#define NHEAD  8
#define DHEAD  64
#define FFDIM  2048
#define NVOCAB 32000
#define NLAYER 4

typedef __attribute__((ext_vector_type(16))) __bf16 v16bf;
typedef __attribute__((ext_vector_type(8)))  __bf16 v8bf;
typedef __attribute__((ext_vector_type(8)))  float  v8f;
typedef __attribute__((ext_vector_type(4)))  int    v4i;

// ---------- helpers ----------
static __device__ __forceinline__ __bf16 f2bf(float f) {
  unsigned u = __builtin_bit_cast(unsigned, f);
  u += 0x7FFFu + ((u >> 16) & 1u);           // round-to-nearest-even
  unsigned short s = (unsigned short)(u >> 16);
  return __builtin_bit_cast(__bf16, s);
}

static __device__ __forceinline__ v16bf ld16(const __bf16* p0, const __bf16* p1) {
  v8bf lo = *(const v8bf*)p0;
  v8bf hi = *(const v8bf*)p1;
  v16bf r;
#pragma unroll
  for (int i = 0; i < 8; ++i) { r[i] = lo[i]; r[i + 8] = hi[i]; }
  return r;
}

// ---------- CDNA5 async global->LDS staging (ASYNCcnt path) ----------
static __device__ __forceinline__ void async_stage16(const __bf16* g, __bf16* l) {
#if __has_builtin(__builtin_amdgcn_global_load_async_to_lds_b128)
  __builtin_amdgcn_global_load_async_to_lds_b128(
      (__attribute__((address_space(1))) v4i*)g,
      (__attribute__((address_space(3))) v4i*)l, 0, 0);
#else
  asm volatile("global_load_async_to_lds_b128 %0, %1, off"
               :: "v"((unsigned)(unsigned long long)l),
                  "v"((unsigned long long)g)
               : "memory");
#endif
}

static __device__ __forceinline__ void wait_async0() {
#if __has_builtin(__builtin_amdgcn_s_wait_asynccnt)
  __builtin_amdgcn_s_wait_asynccnt(0);
#else
  asm volatile("s_wait_asynccnt 0x0" ::: "memory");
#endif
}

// ---------- WMMA GEMM: C[M,N] = A[M,K](bf16,row-major) x Bt[N,K](bf16,row-major)^T ----------
// Block = 8 waves = 128(M) x 64(N) panel; B panel double-buffer staged in LDS via
// global_load_async_to_lds_b128; each wave computes a 16x64 tile (4 accumulators).
// M % 128 == 0, N % 64 == 0, K % 32 == 0, lda/ldb % 8 == 0.
#define BROW 40                                // padded LDS row stride (elems), 16B-aligned, conflict-free
__global__ void __launch_bounds__(256)
gemm_bf16_wmma(const __bf16* __restrict__ A, const __bf16* __restrict__ Bt,
               float* __restrict__ C, int M, int N, int K,
               int lda, int ldb, int ldc) {
  __shared__ __bf16 sB[2][64 * BROW];

  const int lane = threadIdx.x & 31;
  const int wave = threadIdx.x >> 5;
  const int n0 = blockIdx.x << 6;
  const int m0 = (blockIdx.y << 7) + (wave << 4);
  const int l15 = lane & 15, kh = lane >> 4;

  // staging assignment: thread t loads 16B: B row (n0 + t/4), K-chunk (t%4)*8
  const int t = threadIdx.x;
  const int srow = t >> 2, schunk = t & 3;
  const __bf16* sgp = Bt + (size_t)(n0 + srow) * ldb + schunk * 8;
  __bf16* lp0 = &sB[0][srow * BROW + schunk * 8];
  __bf16* lp1 = &sB[1][srow * BROW + schunk * 8];

  // A fragment (ISA 16-bit A layout): row m0+l15; K elems {kh*8+0..7, 16+kh*8+0..7}
  const __bf16* arow = A + (size_t)(m0 + l15) * lda + kh * 8;

  v8f acc0 = {}, acc1 = {}, acc2 = {}, acc3 = {};
  const int nk = K >> 5;

  async_stage16(sgp, lp0);
  for (int i = 0; i < nk; ++i) {
    wait_async0();           // own async stores to LDS complete
    __syncthreads();         // -> everyone's stage for buf[i&1] complete
    if (i + 1 < nk)
      async_stage16(sgp + (size_t)(i + 1) * 32, (i & 1) ? lp0 : lp1);

    const int k0 = i << 5;
    v16bf a = ld16(arow + k0, arow + k0 + 16);

    // B fragment from LDS: column n0+l15 (+16/32/48), K elems {kh*8+0..7, 16+kh*8+0..7}
    const __bf16* bl = &sB[i & 1][l15 * BROW + kh * 8];
    v16bf fb0 = ld16(bl,               bl + 16);
    v16bf fb1 = ld16(bl + 16 * BROW,   bl + 16 * BROW + 16);
    v16bf fb2 = ld16(bl + 32 * BROW,   bl + 32 * BROW + 16);
    v16bf fb3 = ld16(bl + 48 * BROW,   bl + 48 * BROW + 16);

    acc0 = __builtin_amdgcn_wmma_f32_16x16x32_bf16(false, a, false, fb0, (short)0, acc0, false, false);
    acc1 = __builtin_amdgcn_wmma_f32_16x16x32_bf16(false, a, false, fb1, (short)0, acc1, false, false);
    acc2 = __builtin_amdgcn_wmma_f32_16x16x32_bf16(false, a, false, fb2, (short)0, acc2, false, false);
    acc3 = __builtin_amdgcn_wmma_f32_16x16x32_bf16(false, a, false, fb3, (short)0, acc3, false, false);
  }

  // C/D layout: VGPR v -> row m0 + v + 8*kh ; column n0 + l15 (+16/32/48 per subtile)
  float* crow = C + (size_t)(m0 + kh * 8) * ldc + n0 + l15;
#pragma unroll
  for (int v = 0; v < 8; ++v) {
    crow[(size_t)v * ldc +  0] = acc0[v];
    crow[(size_t)v * ldc + 16] = acc1[v];
    crow[(size_t)v * ldc + 32] = acc2[v];
    crow[(size_t)v * ldc + 48] = acc3[v];
  }
}

// ---------- weight transpose + fp32 -> bf16 : in [K,N] -> out [N,K], per z-slice ----------
__global__ void __launch_bounds__(256)
transpose_convert(const float* __restrict__ in, __bf16* __restrict__ out, int K, int N) {
  size_t sl = (size_t)blockIdx.z * (size_t)K * (size_t)N;
  int idx = blockIdx.x * 256 + threadIdx.x;
  if (idx >= K * N) return;
  int k = idx / N, n = idx - k * N;
  out[sl + (size_t)n * K + k] = f2bf(in[sl + idx]);
}

__global__ void __launch_bounds__(256)
convert_bf16(const float* __restrict__ in, __bf16* __restrict__ out, int count) {
  int idx = blockIdx.x * 256 + threadIdx.x;
  if (idx < count) out[idx] = f2bf(in[idx]);
}

// ---------- embedding: h[i,:] = (i%3==0 ? tok[x[i/3]] : reason[i%3-1]) + pos[i/3] ----------
__global__ void __launch_bounds__(256)
embed_kernel(const int* __restrict__ x, const float* __restrict__ tok,
             const float* __restrict__ pos, const float* __restrict__ rt,
             float* __restrict__ h) {
  int i = blockIdx.x;
  int s = i / 3, c = i - s * 3;
  const float* src = (c == 0) ? (tok + (size_t)x[s] * DMODEL) : (rt + (size_t)(c - 1) * DMODEL);
  const float* pp  = pos + (size_t)s * DMODEL;
  for (int d = threadIdx.x; d < DMODEL; d += 256)
    h[(size_t)i * DMODEL + d] = src[d] + pp[d];
}

// ---------- RMSNorm (jnp.linalg.norm flavor) -> bf16 ----------
__global__ void __launch_bounds__(256)
rmsnorm_kernel(const float* __restrict__ h, const float* __restrict__ gamma,
               __bf16* __restrict__ y) {
  __shared__ float red[256];
  int i = blockIdx.x;
  const float* row = h + (size_t)i * DMODEL;
  float s = 0.f;
  for (int d = threadIdx.x; d < DMODEL; d += 256) { float v = row[d]; s += v * v; }
  red[threadIdx.x] = s; __syncthreads();
  for (int off = 128; off > 0; off >>= 1) {
    if (threadIdx.x < off) red[threadIdx.x] += red[threadIdx.x + off];
    __syncthreads();
  }
  float inv = 22.627416997969522f / fmaxf(sqrtf(red[0]), 1e-12f); // sqrt(512)/norm
  for (int d = threadIdx.x; d < DMODEL; d += 256)
    y[(size_t)i * DMODEL + d] = f2bf(row[d] * inv * gamma[d]);
}

// ---------- split qkv[n,1536] -> Q(scaled) [H,n,64], K [H,n,64], V^T [H,64,n] ----------
__global__ void __launch_bounds__(256)
qkv_split(const float* __restrict__ qkv, __bf16* __restrict__ Q,
          __bf16* __restrict__ Kb, __bf16* __restrict__ Vt) {
  int idx = blockIdx.x * 256 + threadIdx.x;
  if (idx >= NTOK * DMODEL) return;
  int i = idx >> 9, col = idx & 511;
  int hh = col >> 6, dh = col & 63;
  const float* rowp = qkv + (size_t)i * (3 * DMODEL);
  Q [((size_t)hh * NTOK + i) * DHEAD + dh] = f2bf(rowp[col] * 0.125f); // 1/sqrt(64)
  Kb[((size_t)hh * NTOK + i) * DHEAD + dh] = f2bf(rowp[DMODEL + col]);
  Vt[(size_t)hh * DHEAD * NTOK + (size_t)dh * NTOK + i] = f2bf(rowp[2 * DMODEL + col]);
}

// ---------- masked softmax row -> bf16 probs ----------
static __device__ __forceinline__ bool allowed_fn(int i, int j, int nsteps) {
  if (j > i) return false;                        // causal
  if ((j % 3) == 0) return true;                  // non-reason key
  int qs = i / 3, ks = j / 3;
  return !((qs > ks) && (qs - nsteps <= ks));     // reason visibility window
}

__global__ void __launch_bounds__(256)
softmax_mask(const float* __restrict__ sim, __bf16* __restrict__ P,
             const int* __restrict__ nsteps_p) {
  __shared__ float red[256];
  const int i = blockIdx.x;
  const int nsteps = nsteps_p[0];
  const float* row = sim + (size_t)i * NTOK;
  float mx = -3.0e38f;
  for (int j = threadIdx.x; j < NTOK; j += 256)
    if (allowed_fn(i, j, nsteps)) mx = fmaxf(mx, row[j]);
  red[threadIdx.x] = mx; __syncthreads();
  for (int off = 128; off > 0; off >>= 1) {
    if (threadIdx.x < off) red[threadIdx.x] = fmaxf(red[threadIdx.x], red[threadIdx.x + off]);
    __syncthreads();
  }
  mx = red[0]; __syncthreads();
  float sum = 0.f;
  for (int j = threadIdx.x; j < NTOK; j += 256)
    if (allowed_fn(i, j, nsteps)) sum += __expf(row[j] - mx);
  red[threadIdx.x] = sum; __syncthreads();
  for (int off = 128; off > 0; off >>= 1) {
    if (threadIdx.x < off) red[threadIdx.x] += red[threadIdx.x + off];
    __syncthreads();
  }
  float inv = 1.f / red[0];
  for (int j = threadIdx.x; j < NTOK; j += 256) {
    float p = allowed_fn(i, j, nsteps) ? __expf(row[j] - mx) * inv : 0.f;
    P[(size_t)i * NTOK + j] = f2bf(p);
  }
}

// ---------- elementwise ----------
__global__ void __launch_bounds__(256)
gelu_bias(const float* __restrict__ in, const float* __restrict__ bias,
          __bf16* __restrict__ out, int count, int cols) {
  int idx = blockIdx.x * 256 + threadIdx.x;
  if (idx >= count) return;
  float v = in[idx] + bias[idx % cols];
  float g = 0.5f * v * (1.f + tanhf(0.7978845608028654f * (v + 0.044715f * v * v * v)));
  out[idx] = f2bf(g);
}

__global__ void __launch_bounds__(256)
add_res(float* __restrict__ h, const float* __restrict__ t, int count) {
  int idx = blockIdx.x * 256 + threadIdx.x;
  if (idx < count) h[idx] += t[idx];
}

__global__ void __launch_bounds__(256)
add_res_bias(float* __restrict__ h, const float* __restrict__ t,
             const float* __restrict__ bias, int count, int cols) {
  int idx = blockIdx.x * 256 + threadIdx.x;
  if (idx < count) h[idx] += t[idx] + bias[idx % cols];
}

// ---------- host ----------
extern "C" void kernel_launch(void* const* d_in, const int* in_sizes, int n_in,
                              void* d_out, int out_size, void* d_ws, size_t ws_size,
                              hipStream_t stream) {
  const int*   x       = (const int*)  d_in[0];
  const float* tok     = (const float*)d_in[1];
  const float* pos     = (const float*)d_in[2];
  const float* rt      = (const float*)d_in[3];
  const float* g_attn  = (const float*)d_in[4];
  const float* wqkv    = (const float*)d_in[5];
  const float* wout    = (const float*)d_in[6];
  const float* g_ff    = (const float*)d_in[7];
  const float* wff1    = (const float*)d_in[8];
  const float* bff1    = (const float*)d_in[9];
  const float* wff2    = (const float*)d_in[10];
  const float* bff2    = (const float*)d_in[11];
  const float* g_final = (const float*)d_in[12];
  const float* wlog    = (const float*)d_in[13];
  const int*   nsteps  = (const int*)  d_in[15];
  float* out = (float*)d_out;

  char* wsp = (char*)d_ws;
  auto alloc = [&](size_t elems, size_t esz) -> void* {
    void* p = (void*)wsp;
    wsp += (elems * esz + 255) & ~(size_t)255;
    return p;
  };

  float*  h     = (float*) alloc((size_t)NTOK * DMODEL, 4);
  __bf16* ybf   = (__bf16*)alloc((size_t)NTOK * DMODEL, 2);
  float*  qkvf  = (float*) alloc((size_t)NTOK * 3 * DMODEL, 4);
  __bf16* Qbf   = (__bf16*)alloc((size_t)NHEAD * NTOK * DHEAD, 2);
  __bf16* Kbf   = (__bf16*)alloc((size_t)NHEAD * NTOK * DHEAD, 2);
  __bf16* Vtbf  = (__bf16*)alloc((size_t)NHEAD * DHEAD * NTOK, 2);
  float*  simf  = (float*) alloc((size_t)NTOK * NTOK, 4);
  __bf16* Pbf   = (__bf16*)alloc((size_t)NTOK * NTOK, 2);
  float*  of    = (float*) alloc((size_t)NTOK * DMODEL, 4);
  __bf16* obf   = (__bf16*)alloc((size_t)NTOK * DMODEL, 2);
  float*  tmpf  = (float*) alloc((size_t)NTOK * DMODEL, 4);
  float*  ff1f  = (float*) alloc((size_t)NTOK * FFDIM, 4);
  __bf16* ubf   = (__bf16*)alloc((size_t)NTOK * FFDIM, 2);
  __bf16* wqkvT = (__bf16*)alloc((size_t)NLAYER * 3 * DMODEL * DMODEL, 2);
  __bf16* woutT = (__bf16*)alloc((size_t)NLAYER * DMODEL * DMODEL, 2);
  __bf16* wff1T = (__bf16*)alloc((size_t)NLAYER * FFDIM * DMODEL, 2);
  __bf16* wff2T = (__bf16*)alloc((size_t)NLAYER * DMODEL * FFDIM, 2);
  __bf16* wlogT = (__bf16*)alloc((size_t)NVOCAB * DMODEL, 2);

  auto gemm = [&](const __bf16* A, const __bf16* Bt, float* C,
                  int M, int N, int K, int lda, int ldb, int ldc) {
    dim3 grid(N / 64, M / 128);
    gemm_bf16_wmma<<<grid, 256, 0, stream>>>(A, Bt, C, M, N, K, lda, ldb, ldc);
  };

  // ---- convert weights to transposed bf16 ([N,K]) ----
  { int kn = DMODEL * 3 * DMODEL;
    transpose_convert<<<dim3((kn + 255) / 256, 1, NLAYER), 256, 0, stream>>>(wqkv, wqkvT, DMODEL, 3 * DMODEL); }
  { int kn = DMODEL * DMODEL;
    transpose_convert<<<dim3((kn + 255) / 256, 1, NLAYER), 256, 0, stream>>>(wout, woutT, DMODEL, DMODEL); }
  { int kn = DMODEL * FFDIM;
    transpose_convert<<<dim3((kn + 255) / 256, 1, NLAYER), 256, 0, stream>>>(wff1, wff1T, DMODEL, FFDIM); }
  { int kn = FFDIM * DMODEL;
    transpose_convert<<<dim3((kn + 255) / 256, 1, NLAYER), 256, 0, stream>>>(wff2, wff2T, FFDIM, DMODEL); }
  { int kn = DMODEL * NVOCAB;
    transpose_convert<<<dim3((kn + 255) / 256, 1, 1), 256, 0, stream>>>(wlog, wlogT, DMODEL, NVOCAB); }

  // ---- embedding ----
  embed_kernel<<<NTOK, 256, 0, stream>>>(x, tok, pos, rt, h);

  const int ND = NTOK * DMODEL, NF = NTOK * FFDIM;
  for (int l = 0; l < NLAYER; ++l) {
    rmsnorm_kernel<<<NTOK, 256, 0, stream>>>(h, g_attn + (size_t)l * DMODEL, ybf);
    gemm(ybf, wqkvT + (size_t)l * 3 * DMODEL * DMODEL, qkvf,
         NTOK, 3 * DMODEL, DMODEL, DMODEL, DMODEL, 3 * DMODEL);
    qkv_split<<<(ND + 255) / 256, 256, 0, stream>>>(qkvf, Qbf, Kbf, Vtbf);

    for (int hh = 0; hh < NHEAD; ++hh) {
      const __bf16* Qh  = Qbf  + (size_t)hh * NTOK * DHEAD;
      const __bf16* Kh  = Kbf  + (size_t)hh * NTOK * DHEAD;
      const __bf16* Vth = Vtbf + (size_t)hh * DHEAD * NTOK;
      gemm(Qh, Kh, simf, NTOK, NTOK, DHEAD, DHEAD, DHEAD, NTOK);        // sim = Q K^T
      softmax_mask<<<NTOK, 256, 0, stream>>>(simf, Pbf, nsteps);
      gemm(Pbf, Vth, of + hh * DHEAD, NTOK, DHEAD, NTOK, NTOK, NTOK, DMODEL); // O_h = P V
    }

    convert_bf16<<<(ND + 255) / 256, 256, 0, stream>>>(of, obf, ND);
    gemm(obf, woutT + (size_t)l * DMODEL * DMODEL, tmpf,
         NTOK, DMODEL, DMODEL, DMODEL, DMODEL, DMODEL);
    add_res<<<(ND + 255) / 256, 256, 0, stream>>>(h, tmpf, ND);

    rmsnorm_kernel<<<NTOK, 256, 0, stream>>>(h, g_ff + (size_t)l * DMODEL, ybf);
    gemm(ybf, wff1T + (size_t)l * FFDIM * DMODEL, ff1f,
         NTOK, FFDIM, DMODEL, DMODEL, DMODEL, FFDIM);
    gelu_bias<<<(NF + 255) / 256, 256, 0, stream>>>(ff1f, bff1 + (size_t)l * FFDIM, ubf, NF, FFDIM);
    gemm(ubf, wff2T + (size_t)l * DMODEL * FFDIM, tmpf,
         NTOK, DMODEL, FFDIM, FFDIM, FFDIM, DMODEL);
    add_res_bias<<<(ND + 255) / 256, 256, 0, stream>>>(h, tmpf, bff2 + (size_t)l * DMODEL, ND, DMODEL);
  }

  rmsnorm_kernel<<<NTOK, 256, 0, stream>>>(h, g_final, ybf);
  gemm(ybf, wlogT, out, NTOK, NVOCAB, DMODEL, DMODEL, DMODEL, NVOCAB);
}
